// AtomicChargeGNN_52673478918910
// MI455X (gfx1250) — matrix operand order repaired and checked by reading, not compiled, not run on hardware
//
#include <hip/hip_runtime.h>

#define NN 100000
#define EE 600000
#define NODE_IN 32
#define EDGE_IN 12
#define HID 128
#define NL 4

typedef __bf16 bf16_t;
typedef __attribute__((ext_vector_type(16))) __bf16 v16bf;
typedef __attribute__((ext_vector_type(8)))  __bf16 v8bf;
typedef __attribute__((ext_vector_type(8)))  float  v8f;

// ---------------------------------------------------------------------------
// WMMA helpers (CDNA5 wave32, V_WMMA_F32_16X16X32_BF16)
// ---------------------------------------------------------------------------
__device__ __forceinline__ v8f wmma_bf16(v16bf a, v16bf b, v8f c) {
    return __builtin_amdgcn_wmma_f32_16x16x32_bf16(
        /*neg_a=*/false, a, /*neg_b=*/false, b,
        /*c_mod=*/(short)0, c, /*reuse_a=*/false, /*reuse_b=*/false);
}

// A-matrix 16x32 bf16 fragment from an LDS row (rowBase = row (lane&15)).
// ISA layout: lane half h=(lane>>4): elems 0..7 -> K = kbase + 8h + i,
//             elems 8..15 -> K = kbase + 16 + 8h + (i-8).
__device__ __forceinline__ v16bf wmma_a_from_lds(const bf16_t* rowBase, int kbase) {
    int half = (threadIdx.x & 31) >> 4;
    const v8bf* p0 = (const v8bf*)(rowBase + kbase + half * 8);
    const v8bf* p1 = (const v8bf*)(rowBase + kbase + 16 + half * 8);
    v8bf lo = *p0, hi = *p1;
    v16bf a;
#pragma unroll
    for (int i = 0; i < 8; ++i) { a[i] = lo[i]; a[i + 8] = hi[i]; }
    return a;
}

// ---------------------------------------------------------------------------
// Weight packing: B-matrix 32x16 bf16 fragment layout.
// ISA layout: lane l holds col N = l&15 ; elem i<8 -> K = (l>>4)*8 + i, i>=8 -> K = 16 + (l>>4)*8 + (i-8).
// Packed as [kt][nt][lane][elem] contiguous (512 bf16 per fragment).
// ---------------------------------------------------------------------------
__device__ __forceinline__ void pack_frag(bf16_t* dst, const float* W,
                                          int Kvalid, int ncols, int ntiles, long r) {
    int  elem = (int)(r & 15);  long r1 = r >> 4;
    int  lane = (int)(r1 & 31); long r2 = r1 >> 5;
    int  nt   = (int)(r2 % ntiles);
    int  kt   = (int)(r2 / ntiles);
    int  K    = kt * 32 + ((elem >> 3) << 4) + ((lane >> 4) << 3) + (elem & 7);
    int  col  = nt * 16 + (lane & 15);
    float v   = (K < Kvalid) ? W[(long)K * ncols + col] : 0.0f;
    dst[r] = (bf16_t)v;
}

__global__ __launch_bounds__(256) void gnn_pack_weights_kernel(
    const float* gW, const float* mW, const float* edW,
    const float* nW, const float* eW, const float* m1W,
    bf16_t* pg, bf16_t* pm, bf16_t* pe, bf16_t* pn, bf16_t* pee, bf16_t* pm1) {
    long t = (long)blockIdx.x * 256 + threadIdx.x;
    const long SG  = (long)NL * 8 * 8 * 512;   // gate / msg packed size
    const long SEe = (long)NL * 12 * 8 * 512;  // edge packed size
    const long SN  = 8L * 512;                 // emb packed size
    const long SM1 = 4L * 4 * 512;             // mlp1 packed size
    if (t < SG) {
        long lsz = 8L * 8 * 512; int l = (int)(t / lsz); long r = t % lsz;
        pack_frag(pg + (long)l * lsz, gW + (long)l * 256 * 128, 256, 128, 8, r);
    } else if ((t -= SG) < SG) {
        long lsz = 8L * 8 * 512; int l = (int)(t / lsz); long r = t % lsz;
        pack_frag(pm + (long)l * lsz, mW + (long)l * 256 * 128, 256, 128, 8, r);
    } else if ((t -= SG) < SEe) {
        long lsz = 12L * 8 * 512; int l = (int)(t / lsz); long r = t % lsz;
        pack_frag(pe + (long)l * lsz, edW + (long)l * 384 * 128, 384, 128, 8, r);
    } else if ((t -= SEe) < SN) {
        pack_frag(pn, nW, NODE_IN, 128, 8, t);
    } else if ((t -= SN) < SN) {
        pack_frag(pee, eW, EDGE_IN, 128, 8, t);   // zero-pad K 12..31
    } else if ((t -= SN) < SM1) {
        pack_frag(pm1, m1W, HID, 64, 4, t);
    }
}

// ---------------------------------------------------------------------------
// Node embedding: h = x @ W + b   (K=32, one WMMA k-step)
// ---------------------------------------------------------------------------
__global__ __launch_bounds__(128) void gnn_node_emb_kernel(
    const float* x, const bf16_t* pw, const float* bias,
    float* h_f32, bf16_t* h_bf16) {
    __shared__ bf16_t xs[64][40];
    int tid = threadIdx.x;
    long nb = (long)blockIdx.x * 64;
    bool full = (nb + 64 <= NN);
    {
        int m = tid >> 1, hh = tid & 1;
        long node = nb + m;
#pragma unroll
        for (int i = 0; i < 16; ++i) {
            float v = (full || node < NN) ? x[node * NODE_IN + hh * 16 + i] : 0.0f;
            xs[m][hh * 16 + i] = (bf16_t)v;
        }
    }
    __syncthreads();
    int lane = tid & 31, wave = tid >> 5;
    const bf16_t* rowBase = &xs[wave * 16 + (lane & 15)][0];
    v16bf a = wmma_a_from_lds(rowBase, 0);
#pragma unroll
    for (int nt = 0; nt < 8; ++nt) {
        v16bf b = ((const v16bf*)(pw + (long)nt * 512))[lane];
        v8f c = {};
        c = wmma_bf16(a, b, c);
        int colc = nt * 16 + (lane & 15);
        float bv = bias[colc];
        if (full) {
#pragma unroll
            for (int r = 0; r < 8; ++r) {
                long node = nb + wave * 16 + r + ((lane >> 4) << 3);
                float v = c[r] + bv;
                h_f32[node * HID + colc]  = v;
                h_bf16[node * HID + colc] = (bf16_t)v;
            }
        } else {
#pragma unroll
            for (int r = 0; r < 8; ++r) {
                long node = nb + wave * 16 + r + ((lane >> 4) << 3);
                if (node < NN) {
                    float v = c[r] + bv;
                    h_f32[node * HID + colc]  = v;
                    h_bf16[node * HID + colc] = (bf16_t)v;
                }
            }
        }
    }
}

// ---------------------------------------------------------------------------
// Edge embedding: e = edge_attr @ W + b  (K=12 zero-padded to 32)
// Grid divides E exactly (600000 % 64 == 0) -> no tail guards.
// ---------------------------------------------------------------------------
__global__ __launch_bounds__(128) void gnn_edge_emb_kernel(
    const float* ea, const bf16_t* pw, const float* bias, bf16_t* e_bf16) {
    __shared__ bf16_t xs[64][40];
    int tid = threadIdx.x;
    long eb = (long)blockIdx.x * 64;
    if (tid < 64) {
        long ed = eb + tid;
#pragma unroll
        for (int k = 0; k < 32; ++k) {
            float v = (k < EDGE_IN) ? ea[ed * EDGE_IN + k] : 0.0f;
            xs[tid][k] = (bf16_t)v;
        }
    }
    __syncthreads();
    int lane = tid & 31, wave = tid >> 5;
    const bf16_t* rowBase = &xs[wave * 16 + (lane & 15)][0];
    v16bf a = wmma_a_from_lds(rowBase, 0);
#pragma unroll
    for (int nt = 0; nt < 8; ++nt) {
        v16bf b = ((const v16bf*)(pw + (long)nt * 512))[lane];
        v8f c = {};
        c = wmma_bf16(a, b, c);
        int colc = nt * 16 + (lane & 15);
        float bv = bias[colc];
#pragma unroll
        for (int r = 0; r < 8; ++r) {
            long ed = eb + wave * 16 + r + ((lane >> 4) << 3);
            e_bf16[ed * HID + colc] = (bf16_t)(c[r] + bv);
        }
    }
}

// ---------------------------------------------------------------------------
// One message-passing layer: 64 edges per block, 256 threads (8 waves).
// Wave w owns N-tile nt=w; B fragments loaded once per (kt,nt) and reused
// across the 4 edge tiles (4x less L2 weight traffic). 8 waves share one
// 52KB LDS staging area -> ~48 resident waves per WGP.
// gate  = sigmoid([x_i,e] @ Wg + bg)          K=256 -> 8 k-steps
// msg   = gate * ([x_j,e] @ Wm + bm)          K=256 -> 8 k-steps
// new_e = [x_j,x_i,e] @ We + be               K=384 -> 12 k-steps
// aggr[col] += msg (atomics);  e += relu(new_e) in place (edge-private rows)
// ---------------------------------------------------------------------------
__global__ __launch_bounds__(256) void gnn_layer_kernel(
    const int* ei, const bf16_t* h_bf, bf16_t* e_bf, float* aggr,
    const bf16_t* pg, const bf16_t* pm, const bf16_t* pe,
    const float* gate_b, const float* msg_b, const float* edge_b, int layer) {
    __shared__ bf16_t xi[64][136];
    __shared__ bf16_t xj[64][136];
    __shared__ bf16_t et[64][136];
    __shared__ int cIdx[64];
    int tid = threadIdx.x;
    long eb = (long)blockIdx.x * 64;
    const int* rowArr = ei;        // edge_index[0] = source (j)
    const int* colArr = ei + EE;   // edge_index[1] = target (i)
    if (tid < 64) cIdx[tid] = colArr[eb + tid];
    {   // gather tiles: 8 threads per edge row, 16 bf16 (32B) per thread
        int m8 = tid >> 3, ch = tid & 7;   // m8: 0..31
#pragma unroll
        for (int q = 0; q < 2; ++q) {
            int m = m8 + q * 32;
            long edge = eb + m;
            int src = rowArr[edge];
            int dst = colArr[edge];
            const uint4* ps = (const uint4*)(h_bf + (long)src * HID + ch * 16);
            const uint4* pd = (const uint4*)(h_bf + (long)dst * HID + ch * 16);
            const uint4* pv = (const uint4*)(e_bf + edge * HID + ch * 16);
            uint4 s0 = ps[0], s1 = ps[1];
            uint4 d0 = pd[0], d1 = pd[1];
            uint4 v0 = pv[0], v1 = pv[1];
            *(uint4*)&xj[m][ch * 16]     = s0;  *(uint4*)&xj[m][ch * 16 + 8] = s1;
            *(uint4*)&xi[m][ch * 16]     = d0;  *(uint4*)&xi[m][ch * 16 + 8] = d1;
            *(uint4*)&et[m][ch * 16]     = v0;  *(uint4*)&et[m][ch * 16 + 8] = v1;
        }
    }
    __syncthreads();

    int lane = tid & 31;
    int nt   = tid >> 5;           // wave id == N-tile id (0..7)
    int rowSel  = lane & 15;
    int colLane = lane & 15;
    const bf16_t* pgL = pg + (long)layer * 8 * 8 * 512;
    const bf16_t* pmL = pm + (long)layer * 8 * 8 * 512;
    const bf16_t* peL = pe + (long)layer * 12 * 8 * 512;

    int colOut = nt * 16 + colLane;
    float gb  = gate_b[layer * HID + colOut];
    float mb  = msg_b[layer * HID + colOut];
    float ebv = edge_b[layer * HID + colOut];

    // ---- gate & msg pass: B loaded once per kt, reused over 4 tiles ----
    v8f gC[4], mC[4];
#pragma unroll
    for (int tile = 0; tile < 4; ++tile) { gC[tile] = (v8f){}; mC[tile] = (v8f){}; }
#pragma unroll
    for (int kt = 0; kt < 8; ++kt) {
        int kb = (kt & 3) * 32;
        v16bf bG = ((const v16bf*)(pgL + ((long)kt * 8 + nt) * 512))[lane];
        v16bf bM = ((const v16bf*)(pmL + ((long)kt * 8 + nt) * 512))[lane];
#pragma unroll
        for (int tile = 0; tile < 4; ++tile) {
            const bf16_t* srcG = (kt < 4) ? &xi[tile * 16 + rowSel][0]
                                          : &et[tile * 16 + rowSel][0];
            const bf16_t* srcM = (kt < 4) ? &xj[tile * 16 + rowSel][0]
                                          : &et[tile * 16 + rowSel][0];
            v16bf aG = wmma_a_from_lds(srcG, kb);
            v16bf aM = wmma_a_from_lds(srcM, kb);
            gC[tile] = wmma_bf16(aG, bG, gC[tile]);
            mC[tile] = wmma_bf16(aM, bM, mC[tile]);
        }
    }
    // finalize gate/msg -> atomic scatter into aggr
#pragma unroll
    for (int tile = 0; tile < 4; ++tile) {
#pragma unroll
        for (int r = 0; r < 8; ++r) {
            int M = tile * 16 + r + ((lane >> 4) << 3);
            float g    = 1.0f / (1.0f + __expf(-(gC[tile][r] + gb)));
            float msgv = g * (mC[tile][r] + mb);
            atomicAdd(&aggr[(long)cIdx[M] * HID + colOut], msgv);
        }
    }

    // ---- edge-update pass ----
    v8f eC[4];
#pragma unroll
    for (int tile = 0; tile < 4; ++tile) eC[tile] = (v8f){};
#pragma unroll
    for (int kt = 0; kt < 12; ++kt) {
        int kb = (kt & 3) * 32;
        v16bf bE = ((const v16bf*)(peL + ((long)kt * 8 + nt) * 512))[lane];
#pragma unroll
        for (int tile = 0; tile < 4; ++tile) {
            const bf16_t* s = (kt < 4) ? &xj[tile * 16 + rowSel][0]
                            : (kt < 8) ? &xi[tile * 16 + rowSel][0]
                                       : &et[tile * 16 + rowSel][0];
            v16bf a = wmma_a_from_lds(s, kb);
            eC[tile] = wmma_bf16(a, bE, eC[tile]);
        }
    }
    // finalize e residual (edge-private rows -> plain stores)
#pragma unroll
    for (int tile = 0; tile < 4; ++tile) {
#pragma unroll
        for (int r = 0; r < 8; ++r) {
            int M = tile * 16 + r + ((lane >> 4) << 3);
            float ne   = eC[tile][r] + ebv;
            float eold = (float)et[M][colOut];
            float enew = eold + (ne > 0.0f ? ne : 0.0f);
            e_bf[(eb + M) * HID + colOut] = (bf16_t)enew;
        }
    }
}

// ---------------------------------------------------------------------------
// Node residual + bf16 mirror, with fused re-zero of aggr for the next layer
// ---------------------------------------------------------------------------
__global__ __launch_bounds__(256) void gnn_node_update_kernel(
    float* h_f32, bf16_t* h_bf16, float* aggr) {
    long i = (long)blockIdx.x * 256 + threadIdx.x;
    if (i < (long)NN * HID) {
        float a = aggr[i];
        float h = h_f32[i] + (a > 0.0f ? a : 0.0f);
        h_f32[i]  = h;
        h_bf16[i] = (bf16_t)h;
        aggr[i]   = 0.0f;       // ready for next layer's scatter
    }
}

__global__ __launch_bounds__(256) void gnn_zero_kernel(float* p, long n) {
    long i = (long)blockIdx.x * 256 + threadIdx.x;
    if (i < n) p[i] = 0.0f;
}

// ---------------------------------------------------------------------------
// Readout: out = relu(h @ W1 + b1) @ W2 + b2  (WMMA for mlp1, shuffle-reduce)
// ---------------------------------------------------------------------------
__global__ __launch_bounds__(128) void gnn_readout_kernel(
    const bf16_t* h_bf, const bf16_t* pw1, const float* b1,
    const float* W2, const float* b2, float* out) {
    __shared__ bf16_t hs[64][136];
    int tid = threadIdx.x;
    long nb = (long)blockIdx.x * 64;
    bool full = (nb + 64 <= NN);
    {
        int m = tid >> 1, hh = tid & 1;
        long node = nb + m;
        uint4 z = {0u, 0u, 0u, 0u};
#pragma unroll
        for (int q = 0; q < 4; ++q) {
            uint4 v = z;
            if (full || node < NN) v = ((const uint4*)(h_bf + node * HID + hh * 64))[q];
            *(uint4*)&hs[m][hh * 64 + q * 8] = v;
        }
    }
    __syncthreads();
    int lane = tid & 31, wave = tid >> 5;
    const bf16_t* rB = &hs[wave * 16 + (lane & 15)][0];
    int colLane = lane & 15;
    float part[8];
#pragma unroll
    for (int r = 0; r < 8; ++r) part[r] = 0.0f;
#pragma unroll
    for (int nt = 0; nt < 4; ++nt) {
        v8f c = {};
#pragma unroll
        for (int kt = 0; kt < 4; ++kt) {
            v16bf a = wmma_a_from_lds(rB, kt * 32);
            v16bf b = ((const v16bf*)(pw1 + ((long)kt * 4 + nt) * 512))[lane];
            c = wmma_bf16(a, b, c);
        }
        int col = nt * 16 + colLane;
        float bv = b1[col], wv = W2[col];
#pragma unroll
        for (int r = 0; r < 8; ++r) {
            float hv = c[r] + bv;
            part[r] += (hv > 0.0f ? hv : 0.0f) * wv;
        }
    }
#pragma unroll
    for (int r = 0; r < 8; ++r) {
        float p = part[r];
#pragma unroll
        for (int m2 = 8; m2 >= 1; m2 >>= 1) p += __shfl_xor(p, m2, 16);
        part[r] = p;
    }
    if (colLane == 0) {
        float bb = b2[0];
#pragma unroll
        for (int r = 0; r < 8; ++r) {
            long node = nb + wave * 16 + r + ((lane >> 4) << 3);
            if (node < NN) out[node] = part[r] + bb;
        }
    }
}

// ---------------------------------------------------------------------------
extern "C" void kernel_launch(void* const* d_in, const int* in_sizes, int n_in,
                              void* d_out, int out_size, void* d_ws, size_t ws_size,
                              hipStream_t stream) {
    (void)in_sizes; (void)n_in; (void)out_size; (void)ws_size;
    const float* x   = (const float*)d_in[0];
    const int*   ei  = (const int*)d_in[1];
    const float* ea  = (const float*)d_in[2];
    const float* nW  = (const float*)d_in[3];
    const float* nb  = (const float*)d_in[4];
    const float* eW  = (const float*)d_in[5];
    const float* ebi = (const float*)d_in[6];
    const float* gW  = (const float*)d_in[7];
    const float* gb  = (const float*)d_in[8];
    const float* mW  = (const float*)d_in[9];
    const float* mb  = (const float*)d_in[10];
    const float* edW = (const float*)d_in[11];
    const float* edb = (const float*)d_in[12];
    const float* m1W = (const float*)d_in[13];
    const float* m1b = (const float*)d_in[14];
    const float* m2W = (const float*)d_in[15];
    const float* m2b = (const float*)d_in[16];
    float* out = (float*)d_out;

    char* ws = (char*)d_ws;
    size_t off = 0;
    auto alloc = [&](size_t bytes) -> char* {
        char* p = ws + off;
        off += (bytes + 255) & ~(size_t)255;
        return p;
    };
    float*  h_f32 = (float*)alloc((size_t)NN * HID * 4);
    bf16_t* h_bf  = (bf16_t*)alloc((size_t)NN * HID * 2);
    bf16_t* e_bf  = (bf16_t*)alloc((size_t)EE * HID * 2);
    float*  aggr  = (float*)alloc((size_t)NN * HID * 4);
    bf16_t* pg    = (bf16_t*)alloc((size_t)NL * 8 * 8 * 512 * 2);
    bf16_t* pm    = (bf16_t*)alloc((size_t)NL * 8 * 8 * 512 * 2);
    bf16_t* pe    = (bf16_t*)alloc((size_t)NL * 12 * 8 * 512 * 2);
    bf16_t* pn    = (bf16_t*)alloc(8 * 512 * 2);
    bf16_t* pee   = (bf16_t*)alloc(8 * 512 * 2);
    bf16_t* pm1   = (bf16_t*)alloc(4 * 4 * 512 * 2);

    // 475136 packed elements total -> 1856 blocks of 256
    gnn_pack_weights_kernel<<<1856, 256, 0, stream>>>(gW, mW, edW, nW, eW, m1W,
                                                      pg, pm, pe, pn, pee, pm1);
    gnn_node_emb_kernel<<<(NN + 63) / 64, 128, 0, stream>>>(x, pn, nb, h_f32, h_bf);
    gnn_edge_emb_kernel<<<EE / 64, 128, 0, stream>>>(ea, pee, ebi, e_bf);

    const long nh = (long)NN * HID;
    gnn_zero_kernel<<<(int)((nh + 255) / 256), 256, 0, stream>>>(aggr, nh);
    for (int l = 0; l < NL; ++l) {
        gnn_layer_kernel<<<EE / 64, 256, 0, stream>>>(ei, h_bf, e_bf, aggr,
                                                      pg, pm, pe, gb, mb, edb, l);
        // h += relu(aggr); mirror to bf16; aggr re-zeroed for next layer
        gnn_node_update_kernel<<<(int)((nh + 255) / 256), 256, 0, stream>>>(h_f32, h_bf, aggr);
    }
    gnn_readout_kernel<<<(NN + 63) / 64, 128, 0, stream>>>(h_bf, pm1, m1b, m2W, m2b, out);
}
